// Attention_43885975831002
// MI455X (gfx1250) — matrix operand order repaired and checked by reading, not compiled, not run on hardware
//
#include <hip/hip_runtime.h>
#include <hip/hip_bf16.h>

// ---------------------------------------------------------------------------
// CDNA5 (gfx1250) WMMA implementation of a BitNet GQA attention block.
// All four matmul stages (Q/K/V projections, QK^T, P*V, output projection)
// use v_wmma_f32_16x16x32_f16. Wave32 operand layouts as loaded here:
//   A (16x32 f16):  lane m=l%16, half h=l/16; a[i]   = A[m][h*16+i]      (i<16) in GEMM/QK^T
//                                             P*V (table order): a[i]=A[m][h*8+i], a[8+i]=A[m][16+h*8+i]
//   B (32x16 f16):  lane n=l%16, half h;      b[i] = B[k][n], k as for a[i] in the same stage
//   C/D (16x16 f32):lane n=l%16, half h;      c[r]   = C[h*8+r][n]
// Trick: computing S^T = K*Q^T makes the softmax->PV A-fragment a pure
// per-lane register conversion (no cross-lane transpose needed).
// ---------------------------------------------------------------------------

typedef __attribute__((ext_vector_type(16))) _Float16 v16h;
typedef __attribute__((ext_vector_type(8)))  _Float16 v8h;
typedef __attribute__((ext_vector_type(8)))  float    v8f;

#define WMMA_F32_F16(Am, Bm, Cm) \
  __builtin_amdgcn_wmma_f32_16x16x32_f16(false, (Am), false, (Bm), (short)0, (Cm), false, false)

union AFrag { v16h v; v8h h[2]; };

constexpr int kB = 2, kT = 2048, kD = 1024, kH = 16, kKVH = 4, kHD = 64, kNRep = 4;
constexpr float kSoftcap = 50.0f;
constexpr float kEps = 1e-6f;

// ---------------- alpha = mean(|w|), single block => deterministic ----------
__global__ __launch_bounds__(256) void absmean_kernel(const float* __restrict__ w, int n,
                                                      float* __restrict__ out) {
  __shared__ float sm[256];
  float s = 0.f;
  for (int i = threadIdx.x; i < n; i += 256) s += fabsf(w[i]);
  sm[threadIdx.x] = s;
  __syncthreads();
  for (int off = 128; off > 0; off >>= 1) {
    if ((int)threadIdx.x < off) sm[threadIdx.x] += sm[threadIdx.x + off];
    __syncthreads();
  }
  if (threadIdx.x == 0) out[0] = sm[0] / (float)n;
}

// ---------------- sign(w) -> f16 {-1,0,+1} ---------------------------------
__global__ void pack_sign_kernel(const float* __restrict__ w, _Float16* __restrict__ o, int n) {
  int i = blockIdx.x * 256 + threadIdx.x;
  if (i < n) {
    float x = w[i];
    o[i] = (_Float16)((x > 0.f) ? 1.f : ((x < 0.f) ? -1.f : 0.f));
  }
}

__global__ void cvt_f16_kernel(const float* __restrict__ s, _Float16* __restrict__ d, int n) {
  int i = blockIdx.x * 256 + threadIdx.x;
  if (i < n) d[i] = (_Float16)s[i];
}

// ---------------- WMMA GEMM: C[M,N] = alpha * A[M,K] @ Bw[N,K]^T ------------
// One wave computes a 16(M) x 64(N) tile; A fragment reused across 4 B tiles.
__global__ __launch_bounds__(256) void gemm_wmma_kernel(
    const _Float16* __restrict__ A, const _Float16* __restrict__ Bw,
    float* __restrict__ C, const float* __restrict__ alpha_ptr,
    int M, int N, int K) {
  const int lane = threadIdx.x & 31;
  const int wave = (blockIdx.x * 256 + threadIdx.x) >> 5;
  const int mt = M >> 4, nt = N >> 6;
  if (wave >= mt * nt) return;
  const int tm = wave % mt, tn = wave / mt;
  const int row0 = tm << 4, col0 = tn << 6;
  const int m = lane & 15, hf = lane >> 4;

  v8f acc0 = {}, acc1 = {}, acc2 = {}, acc3 = {};
  const _Float16* arow = A + (size_t)(row0 + m) * K;
  const _Float16* b0p = Bw + (size_t)(col0 + 0  + m) * K;
  const _Float16* b1p = Bw + (size_t)(col0 + 16 + m) * K;
  const _Float16* b2p = Bw + (size_t)(col0 + 32 + m) * K;
  const _Float16* b3p = Bw + (size_t)(col0 + 48 + m) * K;

  for (int k0 = 0; k0 < K; k0 += 32) {
    AFrag a;
    a.h[0] = *(const v8h*)(arow + k0 + hf * 16);
    a.h[1] = *(const v8h*)(arow + k0 + hf * 16 + 8);
    const int kb = k0 + hf * 16;
    v16h b0 = *(const v16h*)(b0p + kb);
    v16h b1 = *(const v16h*)(b1p + kb);
    v16h b2 = *(const v16h*)(b2p + kb);
    v16h b3 = *(const v16h*)(b3p + kb);
    acc0 = WMMA_F32_F16(a.v, b0, acc0);
    acc1 = WMMA_F32_F16(a.v, b1, acc1);
    acc2 = WMMA_F32_F16(a.v, b2, acc2);
    acc3 = WMMA_F32_F16(a.v, b3, acc3);
  }
  const float alpha = alpha_ptr[0];
  for (int r = 0; r < 8; ++r) {
    float* crow = C + (size_t)(row0 + hf * 8 + r) * N + col0 + m;
    crow[0]  = acc0[r] * alpha;
    crow[16] = acc1[r] * alpha;
    crow[32] = acc2[r] * alpha;
    crow[48] = acc3[r] * alpha;
  }
}

// ---------------- RMSNorm + RoPE, one wave per (b, head, t) row of 64 -------
__global__ __launch_bounds__(256) void qk_prep_kernel(
    const float* __restrict__ raw,      // [B][T][NH*64]
    _Float16* __restrict__ outp,        // [B][NH][T][64]
    const float* __restrict__ nw, const float* __restrict__ cosp,
    const float* __restrict__ sinp, int NH, int nwaves) {
  int wave = (blockIdx.x * 256 + threadIdx.x) >> 5;
  int lane = threadIdx.x & 31;
  if (wave >= nwaves) return;
  int t = wave % kT;
  int rem = wave / kT;
  int nh = rem % NH;
  int b = rem / NH;
  const float* rb = raw + ((size_t)b * kT + t) * ((size_t)NH * kHD) + (size_t)nh * kHD;
  float x0 = rb[lane], x1 = rb[lane + 32];
  float ss = x0 * x0 + x1 * x1;
  ss += __shfl_xor(ss, 16);
  ss += __shfl_xor(ss, 8);
  ss += __shfl_xor(ss, 4);
  ss += __shfl_xor(ss, 2);
  ss += __shfl_xor(ss, 1);
  float inv = 1.f / sqrtf(ss * (1.f / (float)kHD) + kEps);
  float y0 = x0 * inv * nw[lane];
  float y1 = x1 * inv * nw[lane + 32];
  const float* ct = cosp + (size_t)t * kHD;
  const float* st = sinp + (size_t)t * kHD;
  float r0 = y0 * ct[lane]      - y1 * st[lane];       // d < 32: rot = -x2
  float r1 = y1 * ct[lane + 32] + y0 * st[lane + 32];  // d >=32: rot = +x1
  _Float16* ob = outp + (((size_t)b * NH + nh) * kT + t) * kHD;
  ob[lane]      = (_Float16)r0;
  ob[lane + 32] = (_Float16)r1;
}

// ---------------- V: f32 [B][T][KVH*64] -> f16 transposed [B][KVH][64][T] ---
__global__ void v_prep_kernel(const float* __restrict__ vraw, _Float16* __restrict__ vT, int n) {
  int i = blockIdx.x * 256 + threadIdx.x;
  if (i >= n) return;
  int t  = i % kT;
  int d  = (i / kT) % kHD;
  int kv = (i / (kT * kHD)) % kKVH;
  int b  = i / (kT * kHD * kKVH);
  vT[i] = (_Float16)vraw[((size_t)b * kT + t) * (kKVH * kHD) + kv * kHD + d];
}

// ---------------- Flash attention with tanh softcap, causal, online max ----
// One wave: 16 queries x full head dim (64). S^T = K*Q^T so softmax->PV
// fragment conversion is per-lane only.
__global__ __launch_bounds__(256) void attn_kernel(
    const _Float16* __restrict__ q16,   // [B][H][T][64]
    const _Float16* __restrict__ k16,   // [B][KVH][T][64]
    const _Float16* __restrict__ vT,    // [B][KVH][64][T]
    _Float16* __restrict__ ctx,         // [B][T][H*64]
    int nwaves) {
  int wave = (blockIdx.x * 256 + threadIdx.x) >> 5;
  int lane = threadIdx.x & 31;
  if (wave >= nwaves) return;
  const int QT = kT / 16;
  int qt = wave % QT;
  int bh = wave / QT;
  int hh = bh % kH;
  int b  = bh / kH;
  int kv = hh / kNRep;
  int row0 = qt * 16;
  int m = lane & 15, hf = lane >> 4;

  const _Float16* qb  = q16 + (((size_t)b * kH + hh) * kT) * kHD;
  const _Float16* kbp = k16 + (((size_t)b * kKVH + kv) * kT) * kHD;
  const _Float16* vbp = vT  + (((size_t)b * kKVH + kv) * kHD) * kT;

  // Q as B-operand (K-dim = head dim), preloaded for both k-steps.
  v16h qB0 = *(const v16h*)(qb + (size_t)(row0 + m) * kHD + hf * 16);
  v16h qB1 = *(const v16h*)(qb + (size_t)(row0 + m) * kHD + 32 + hf * 16);

  v8f o0 = {}, o1 = {}, o2 = {}, o3 = {};
  float lsum = 0.f, mrow = -1e30f;
  const int tglob = row0 + m;
  const int smax = row0 + 16;  // causal: keys s <= max query in tile

  for (int s0 = 0; s0 < smax; s0 += 32) {
    float cbuf[16];
    for (int u = 0; u < 2; ++u) {
      const _Float16* krow = kbp + (size_t)(s0 + u * 16 + m) * kHD;
      AFrag ka;
      ka.h[0] = *(const v8h*)(krow + hf * 16);
      ka.h[1] = *(const v8h*)(krow + hf * 16 + 8);
      v8f st = {};
      st = WMMA_F32_F16(ka.v, qB0, st);
      ka.h[0] = *(const v8h*)(krow + 32 + hf * 16);
      ka.h[1] = *(const v8h*)(krow + 32 + hf * 16 + 8);
      st = WMMA_F32_F16(ka.v, qB1, st);
      const int sbase = s0 + u * 16 + hf * 8;
      for (int r = 0; r < 8; ++r) {
        float sv = st[r] * 0.125f;  // 1/sqrt(64)
        float cap = tanhf(sv * (1.0f / kSoftcap)) * kSoftcap;
        cbuf[u * 8 + r] = (sbase + r <= tglob) ? cap : -1e30f;
      }
    }
    // online softmax (row stats live in lanes l and l^16 for column t=l%16)
    float mt = cbuf[0];
#pragma unroll
    for (int i = 1; i < 16; ++i) mt = fmaxf(mt, cbuf[i]);
    mt = fmaxf(mt, __shfl_xor(mt, 16));
    float mnew = fmaxf(mrow, mt);
    float corr = __expf(mrow - mnew);
    mrow = mnew;
    lsum *= corr;
    AFrag pa;
#pragma unroll
    for (int i = 0; i < 16; ++i) {
      float p = __expf(cbuf[i] - mnew);
      lsum += p;
      pa.v[i] = (_Float16)p;  // S^T C-layout == P A-layout, per-lane only
    }
    // rescale O accumulators (t = hf*8+r lives in lane hf*8+r)
#pragma unroll
    for (int r = 0; r < 8; ++r) {
      float cr = __shfl(corr, hf * 8 + r);
      o0[r] *= cr; o1[r] *= cr; o2[r] *= cr; o3[r] *= cr;
    }
    const int kcol = s0 + hf * 8;
    v16h vB0 = __builtin_shufflevector(*(const v8h*)(vbp + (size_t)(0  + m) * kT + kcol), *(const v8h*)(vbp + (size_t)(0  + m) * kT + kcol + 16), 0,1,2,3,4,5,6,7,8,9,10,11,12,13,14,15);
    v16h vB1 = __builtin_shufflevector(*(const v8h*)(vbp + (size_t)(16 + m) * kT + kcol), *(const v8h*)(vbp + (size_t)(16 + m) * kT + kcol + 16), 0,1,2,3,4,5,6,7,8,9,10,11,12,13,14,15);
    v16h vB2 = __builtin_shufflevector(*(const v8h*)(vbp + (size_t)(32 + m) * kT + kcol), *(const v8h*)(vbp + (size_t)(32 + m) * kT + kcol + 16), 0,1,2,3,4,5,6,7,8,9,10,11,12,13,14,15);
    v16h vB3 = __builtin_shufflevector(*(const v8h*)(vbp + (size_t)(48 + m) * kT + kcol), *(const v8h*)(vbp + (size_t)(48 + m) * kT + kcol + 16), 0,1,2,3,4,5,6,7,8,9,10,11,12,13,14,15);
    o0 = WMMA_F32_F16(pa.v, vB0, o0);
    o1 = WMMA_F32_F16(pa.v, vB1, o1);
    o2 = WMMA_F32_F16(pa.v, vB2, o2);
    o3 = WMMA_F32_F16(pa.v, vB3, o3);
  }

  lsum += __shfl_xor(lsum, 16);
  float invl = 1.f / lsum;
#pragma unroll
  for (int r = 0; r < 8; ++r) {
    float wsc = __shfl(invl, hf * 8 + r);
    _Float16* crow = ctx + ((size_t)b * kT + (size_t)(row0 + hf * 8 + r)) * ((size_t)kH * kHD)
                     + (size_t)hh * kHD + m;
    crow[0]  = (_Float16)(o0[r] * wsc);
    crow[16] = (_Float16)(o1[r] * wsc);
    crow[32] = (_Float16)(o2[r] * wsc);
    crow[48] = (_Float16)(o3[r] * wsc);
  }
}

// ---------------------------------------------------------------------------
extern "C" void kernel_launch(void* const* d_in, const int* in_sizes, int n_in,
                              void* d_out, int out_size, void* d_ws, size_t ws_size,
                              hipStream_t stream) {
  (void)in_sizes; (void)n_in; (void)out_size; (void)ws_size;
  const float* x    = (const float*)d_in[0];
  const float* cosp = (const float*)d_in[1];
  const float* sinp = (const float*)d_in[2];
  // d_in[3] = mask: causality handled analytically in attn_kernel
  const float* wq   = (const float*)d_in[4];
  const float* wk   = (const float*)d_in[5];
  const float* wv   = (const float*)d_in[6];
  const float* wo   = (const float*)d_in[7];
  const float* qnw  = (const float*)d_in[8];
  const float* knw  = (const float*)d_in[9];
  float* out = (float*)d_out;
  char* ws = (char*)d_ws;

  constexpr size_t nX  = (size_t)kB * kT * kD;          // 4Mi elements
  constexpr size_t nWQ = (size_t)kD * kD;               // 1Mi
  constexpr size_t nWK = (size_t)(kKVH * kHD) * kD;     // 256Ki
  constexpr size_t nKV = (size_t)kB * kT * kKVH * kHD;  // 1Mi

  size_t off = 0;
  const size_t off_alpha = off; off += 256;
  const size_t off_x16   = off; off += nX * 2;   // later reused as q16
  const size_t off_wq16  = off; off += nWQ * 2;
  const size_t off_wk16  = off; off += nWK * 2;
  const size_t off_wv16  = off; off += nWK * 2;
  const size_t off_wo16  = off; off += nWQ * 2;
  const size_t off_qraw  = off; off += nX * 4;   // later reused as ctx16
  const size_t off_kraw  = off; off += nKV * 4;
  const size_t off_vraw  = off; off += nKV * 4;  // later reused as k16
  const size_t off_vT16  = off; off += nKV * 2;  // ~39 MB total

  float*    alphas = (float*)(ws + off_alpha);
  _Float16* x16    = (_Float16*)(ws + off_x16);
  _Float16* wq16   = (_Float16*)(ws + off_wq16);
  _Float16* wk16   = (_Float16*)(ws + off_wk16);
  _Float16* wv16   = (_Float16*)(ws + off_wv16);
  _Float16* wo16   = (_Float16*)(ws + off_wo16);
  float*    qraw   = (float*)(ws + off_qraw);
  float*    kraw   = (float*)(ws + off_kraw);
  float*    vraw   = (float*)(ws + off_vraw);
  _Float16* vT16   = (_Float16*)(ws + off_vT16);
  _Float16* q16    = (_Float16*)(ws + off_x16);   // alias: x16 dead after GEMMs
  _Float16* k16    = (_Float16*)(ws + off_vraw);  // alias: vraw dead after v_prep
  _Float16* ctx16  = (_Float16*)(ws + off_qraw);  // alias: qraw dead after q_prep

  // 1) per-weight alpha = mean(|w|)
  absmean_kernel<<<1, 256, 0, stream>>>(wq, (int)nWQ, alphas + 0);
  absmean_kernel<<<1, 256, 0, stream>>>(wk, (int)nWK, alphas + 1);
  absmean_kernel<<<1, 256, 0, stream>>>(wv, (int)nWK, alphas + 2);
  absmean_kernel<<<1, 256, 0, stream>>>(wo, (int)nWQ, alphas + 3);

  // 2) pack sign weights + activations to f16
  pack_sign_kernel<<<(int)((nWQ + 255) / 256), 256, 0, stream>>>(wq, wq16, (int)nWQ);
  pack_sign_kernel<<<(int)((nWK + 255) / 256), 256, 0, stream>>>(wk, wk16, (int)nWK);
  pack_sign_kernel<<<(int)((nWK + 255) / 256), 256, 0, stream>>>(wv, wv16, (int)nWK);
  pack_sign_kernel<<<(int)((nWQ + 255) / 256), 256, 0, stream>>>(wo, wo16, (int)nWQ);
  cvt_f16_kernel<<<(int)((nX + 255) / 256), 256, 0, stream>>>(x, x16, (int)nX);

  // 3) projections (WMMA)
  const int Mrows = kB * kT;
  {
    int waves = (Mrows / 16) * (kD / 64);
    gemm_wmma_kernel<<<(waves + 7) / 8, 256, 0, stream>>>(x16, wq16, qraw, alphas + 0, Mrows, kD, kD);
  }
  {
    int waves = (Mrows / 16) * ((kKVH * kHD) / 64);
    gemm_wmma_kernel<<<(waves + 7) / 8, 256, 0, stream>>>(x16, wk16, kraw, alphas + 1, Mrows, kKVH * kHD, kD);
    gemm_wmma_kernel<<<(waves + 7) / 8, 256, 0, stream>>>(x16, wv16, vraw, alphas + 2, Mrows, kKVH * kHD, kD);
  }

  // 4) V transpose+cvt, then K/Q RMSNorm+RoPE (order matters for aliasing)
  {
    int n = (int)(nKV);
    v_prep_kernel<<<(n + 255) / 256, 256, 0, stream>>>(vraw, vT16, n);
  }
  {
    int nwaves = kB * kKVH * kT;
    qk_prep_kernel<<<(nwaves * 32 + 255) / 256, 256, 0, stream>>>(kraw, k16, knw, cosp, sinp, kKVH, nwaves);
  }
  {
    int nwaves = kB * kH * kT;
    qk_prep_kernel<<<(nwaves * 32 + 255) / 256, 256, 0, stream>>>(qraw, q16, qnw, cosp, sinp, kH, nwaves);
  }

  // 5) flash attention (WMMA)
  {
    int nwaves = kB * kH * (kT / 16);
    attn_kernel<<<(nwaves * 32 + 255) / 256, 256, 0, stream>>>(q16, k16, vT16, ctx16, nwaves);
  }

  // 6) output projection straight into d_out (WMMA)
  {
    int waves = (Mrows / 16) * (kD / 64);
    gemm_wmma_kernel<<<(waves + 7) / 8, 256, 0, stream>>>(ctx16, wo16, out, alphas + 3, Mrows, kD, kD);
  }
}
